// Head_10230612099156
// MI455X (gfx1250) — matrix operand order repaired and checked by reading, not compiled, not run on hardware
//
#include <hip/hip_runtime.h>
#include <hip/hip_bf16.h>

// Problem constants (reference: B=4, T=4096, C=512, H=64)
#define BATCH 4
#define TLEN  4096
#define CDIM  512
#define HDIM  64

typedef __attribute__((ext_vector_type(16))) _Float16 v16h;
typedef __attribute__((ext_vector_type(8)))  float    v8f;
typedef __attribute__((ext_vector_type(4)))  int      v4i;

union AU { v16h h; unsigned int u[8]; _Float16 e[16]; };

#if defined(__has_builtin)
#if __has_builtin(__builtin_amdgcn_global_load_async_to_lds_b128) && \
    __has_builtin(__builtin_amdgcn_s_wait_asynccnt)
#define HAVE_ASYNC_LDS 1
#endif
#endif

// async-to-LDS builtin takes (v4i addrspace(1)* src, v4i addrspace(3)* dst,
// imm offset, imm cpol) -- per hipcc diagnostic on round 2.
#define AS1V(p) ((__attribute__((address_space(1))) v4i*)(p))
#define AS3V(p) ((__attribute__((address_space(3))) v4i*)(p))

// A-matrix (16x32 f16) pair index for VGPR j given lane-half sel.
// Contiguous K order: lanes0-15: V0..7 K=0..15; lanes16-31: K=16..31.
__device__ __forceinline__ int a_pair_idx(int j, int sel) {
    return sel * 8 + j;
}

// -------------------------------------------------------------------------
// Kernel 1: convert Wk/Wq/Wv (fp32 [C,H]) into f16 packed in the 16-bit
// B-matrix (32x16) lane layout, this file's contiguous-K order: VGPR j of lane (n,sel) = rows sel*16+2j,+1
// of the K-chunk, column n. Stored as [matrix][kc(16)][nt(4)][lane(32)][j(8)]
// -------------------------------------------------------------------------
__global__ __launch_bounds__(128) void head_wconv(
    const float* __restrict__ Wk, const float* __restrict__ Wq,
    const float* __restrict__ Wv, unsigned int* __restrict__ Wc)
{
    const int m    = blockIdx.y;            // matrix 0..2
    const int kc   = blockIdx.x;            // K-chunk 0..15 (32 rows each)
    const int nt   = threadIdx.x >> 5;      // N-tile 0..3
    const int lane = threadIdx.x & 31;
    const int n = lane & 15, sel = lane >> 4;
    const float* W = (m == 0) ? Wk : (m == 1) ? Wq : Wv;
    unsigned int* dst = Wc + (((m * 16 + kc) * 4 + nt) * 32 + lane) * 8;
    const int col = nt * 16 + n;
    for (int j = 0; j < 8; ++j) {
        const int row = kc * 32 + sel * 16 + 2 * j;
        _Float16 h0 = (_Float16)W[row * HDIM + col];
        _Float16 h1 = (_Float16)W[(row + 1) * HDIM + col];
        unsigned short u0 = __builtin_bit_cast(unsigned short, h0);
        unsigned short u1 = __builtin_bit_cast(unsigned short, h1);
        dst[j] = (unsigned int)u0 | ((unsigned int)u1 << 16);
    }
}

// -------------------------------------------------------------------------
// Kernel 2: projections via WMMA. One wave per 16 rows of x.
//   Qr = x@Wk (query role, reference computes k@q^T), Kr = x@Wq,
//   Vt = (x@Wv)^T stored [B][H][T] f16 so attention V B-tiles are contiguous.
// -------------------------------------------------------------------------
__global__ __launch_bounds__(32) void head_proj(
    const float* __restrict__ x, const unsigned int* __restrict__ Wc,
    _Float16* __restrict__ Qr, _Float16* __restrict__ Kr,
    _Float16* __restrict__ Vt)
{
    const int lane = threadIdx.x;
    const int n = lane & 15, sel = lane >> 4;
    const size_t row0 = (size_t)blockIdx.x * 16;

    v8f acc[3][4];
    for (int m = 0; m < 3; ++m)
        for (int t = 0; t < 4; ++t) acc[m][t] = (v8f){};

    const float* xrow = x + (row0 + (size_t)n) * CDIM;  // A row = lane%16

    for (int kc = 0; kc < 16; ++kc) {                    // K = 512 = 16*32
        AU aX;
        for (int j = 0; j < 8; ++j) {                    // fp32 -> f16 A tile
            const int pj = a_pair_idx(j, sel);
            aX.e[2 * j]     = (_Float16)xrow[kc * 32 + 2 * pj];
            aX.e[2 * j + 1] = (_Float16)xrow[kc * 32 + 2 * pj + 1];
        }
        for (int m = 0; m < 3; ++m) {
            for (int nt = 0; nt < 4; ++nt) {
                AU bW;
                const unsigned int* p =
                    Wc + (((m * 16 + kc) * 4 + nt) * 32 + lane) * 8;
                for (int j = 0; j < 8; ++j) bW.u[j] = p[j];
                acc[m][nt] = __builtin_amdgcn_wmma_f32_16x16x32_f16(
                    false, aX.h, false, bW.h, (short)0, acc[m][nt],
                    false, false);
            }
        }
    }
    // C-layout: VGPR r -> row r (lanes 0-15) / r+8 (lanes 16-31), col = n.
    for (int r = 0; r < 8; ++r) {
        const size_t row = row0 + (size_t)(r + sel * 8);
        for (int nt = 0; nt < 4; ++nt) {
            Qr[row * HDIM + nt * 16 + n] = (_Float16)acc[0][nt][r];
            Kr[row * HDIM + nt * 16 + n] = (_Float16)acc[1][nt][r];
        }
        const int b = (int)(row >> 12), t = (int)(row & (TLEN - 1));
        for (int nt = 0; nt < 4; ++nt)
            Vt[((size_t)b * HDIM + nt * 16 + n) * TLEN + t] =
                (_Float16)acc[2][nt][r];
    }
}

// -------------------------------------------------------------------------
// Kernel 3: causal flash attention. One wave per (batch, 16-row tile).
// KV tiles of 32 columns (WMMA K=32). 8 WMMAs / iteration.
// Async double-buffered K/V staging in LDS when the toolchain exposes
// GLOBAL_LOAD_ASYNC_TO_LDS; otherwise direct global->VGPR loads.
// -------------------------------------------------------------------------
__global__ __launch_bounds__(32) void head_attn(
    const _Float16* __restrict__ Qr, const _Float16* __restrict__ Kr,
    const _Float16* __restrict__ Vt, float* __restrict__ out)
{
    __shared__ unsigned int pLdsU[16 * 32 / 2];          // 16x32 f16 P tile
    _Float16* pLds = (_Float16*)pLdsU;
#ifdef HAVE_ASYNC_LDS
    __shared__ _Float16 kTile[2][32 * HDIM];             // [rows 32][h 64]
    __shared__ _Float16 vTile[2][HDIM * 32];             // [h 64][t 32]
#endif

    const int lane = threadIdx.x;
    const int n = lane & 15, sel = lane >> 4;
    const int b = blockIdx.y;
    const int rowbase = blockIdx.x * 16;

    const _Float16* Qb = Qr + (size_t)b * TLEN * HDIM;
    const _Float16* Kb = Kr + (size_t)b * TLEN * HDIM;
    const _Float16* Vb = Vt + (size_t)b * HDIM * TLEN;

    // Persistent A operand: query tile 16x64 as two 16x32 A tiles.
    AU aQ[2];
    {
        const unsigned int* qp =
            (const unsigned int*)(Qb + (size_t)(rowbase + n) * HDIM);
        for (int hc = 0; hc < 2; ++hc)
            for (int j = 0; j < 8; ++j)
                aQ[hc].u[j] = qp[hc * 16 + a_pair_idx(j, sel)];
    }

    v8f acc0 = (v8f){}, acc1 = (v8f){}, acc2 = (v8f){}, acc3 = (v8f){};
    float mrow[8], lrow[8];
    for (int r = 0; r < 8; ++r) { mrow[r] = -3.0e38f; lrow[r] = 0.0f; }

    // score scale C^-0.5 folded with log2(e) so we can use exp2
    const float kScale = 0.04419417382415922f * 1.4426950408889634f;

    const int ntiles = (rowbase + 15) / 32 + 1;

#ifdef HAVE_ASYNC_LDS
    // Issue one kv tile's 16 async b128 transfers (8 for K, 8 for V).
    // K tile: rows s0..s0+31 are 4KB contiguous in Kb (row-major, 128B/row).
    // V tile: 64 h-rows x 64B from transposed V (row stride TLEN halves).
    const int vh  = lane >> 2;          // base h-row handled by this lane
    const int vsg = lane & 3;           // 16B segment within 64B h-row
#define ISSUE_TILE(s0_, buf_)                                                \
    {                                                                        \
        const char* gk = (const char*)(Kb + (size_t)(s0_) * HDIM) + lane*16; \
        char*       lk = (char*)&kTile[(buf_)][0] + lane * 16;               \
        _Pragma("unroll")                                                    \
        for (int st = 0; st < 8; ++st)                                       \
            __builtin_amdgcn_global_load_async_to_lds_b128(                  \
                AS1V(gk + st * 512), AS3V(lk + st * 512), 0, 0);             \
        const char* gv = (const char*)(Vb + (size_t)vh * TLEN + (s0_)) +     \
                         vsg * 16;                                           \
        char*       lv = (char*)&vTile[(buf_)][0] + vh * 64 + vsg * 16;      \
        _Pragma("unroll")                                                    \
        for (int st = 0; st < 8; ++st)                                       \
            __builtin_amdgcn_global_load_async_to_lds_b128(                  \
                AS1V(gv + (size_t)st * 8 * TLEN * 2),                        \
                AS3V(lv + st * 8 * 64), 0, 0);                               \
    }
    ISSUE_TILE(0, 0)
#endif

    for (int jt = 0; jt < ntiles; ++jt) {
        const int s0 = jt * 32;
        AU bK[2][2], bV[4];

#ifdef HAVE_ASYNC_LDS
        const int buf = jt & 1;
        if (jt + 1 < ntiles) {
            ISSUE_TILE(s0 + 32, buf ^ 1)                  // prefetch next tile
            __builtin_amdgcn_s_wait_asynccnt(16);         // current tile done
        } else {
            __builtin_amdgcn_s_wait_asynccnt(0);
        }
        __syncthreads();
        // Score B tiles from LDS: 16 contiguous halves per lane.
        for (int nh = 0; nh < 2; ++nh)
            for (int hc = 0; hc < 2; ++hc) {
                const unsigned int* p = (const unsigned int*)
                    (&kTile[buf][(nh * 16 + n) * HDIM + hc * 32 + sel * 16]);
                for (int j = 0; j < 8; ++j) bK[nh][hc].u[j] = p[j];
            }
#else
        if (jt + 1 < ntiles) {                           // prefetch next tile
            __builtin_prefetch(Kb + (size_t)(s0 + 32 + n) * HDIM, 0, 0);
            __builtin_prefetch(Vb + (size_t)n * TLEN + s0 + 32, 0, 0);
        }
        // B tiles for S = Q @ K^T : element = Kb[s0+nh*16+n][hc*32+kk].
        for (int nh = 0; nh < 2; ++nh) {
            const unsigned int* kp =
                (const unsigned int*)(Kb + (size_t)(s0 + nh * 16 + n) * HDIM);
            for (int hc = 0; hc < 2; ++hc) {
                const unsigned int* p = kp + hc * 16 + sel * 8;
                for (int j = 0; j < 8; ++j) bK[nh][hc].u[j] = p[j];
            }
        }
#endif
        v8f s0t = (v8f){}, s1t = (v8f){};
        s0t = __builtin_amdgcn_wmma_f32_16x16x32_f16(false, aQ[0].h, false, bK[0][0].h, (short)0, s0t, false, false);
        s0t = __builtin_amdgcn_wmma_f32_16x16x32_f16(false, aQ[1].h, false, bK[0][1].h, (short)0, s0t, false, false);
        s1t = __builtin_amdgcn_wmma_f32_16x16x32_f16(false, aQ[0].h, false, bK[1][0].h, (short)0, s1t, false, false);
        s1t = __builtin_amdgcn_wmma_f32_16x16x32_f16(false, aQ[1].h, false, bK[1][1].h, (short)0, s1t, false, false);

        // Causal mask + online softmax (row reductions inside 16-lane halves)
        float fac[8];
        const int col0 = s0 + n, col1 = s0 + 16 + n;
        for (int r = 0; r < 8; ++r) {
            const int row = rowbase + r + sel * 8;
            float v0 = (col0 <= row) ? s0t[r] * kScale : -3.0e38f;
            float v1 = (col1 <= row) ? s1t[r] * kScale : -3.0e38f;
            float mx = fmaxf(v0, v1);
            for (int off = 1; off < 16; off <<= 1)
                mx = fmaxf(mx, __shfl_xor(mx, off, 32));
            const float mnew = fmaxf(mrow[r], mx);
            fac[r] = exp2f(mrow[r] - mnew);
            mrow[r] = mnew;
            v0 = exp2f(v0 - mnew);
            v1 = exp2f(v1 - mnew);
            float rs = v0 + v1;
            for (int off = 1; off < 16; off <<= 1)
                rs += __shfl_xor(rs, off, 32);
            lrow[r] = lrow[r] * fac[r] + rs;
            s0t[r] = v0; s1t[r] = v1;
        }

        // C-layout f32 P -> LDS f16 row-major -> reload in A(16x32) layout
        for (int r = 0; r < 8; ++r) {
            const int row = r + sel * 8;
            pLds[row * 32 + n]      = (_Float16)s0t[r];
            pLds[row * 32 + 16 + n] = (_Float16)s1t[r];
        }
        __syncthreads();
        AU aP;
        {
            const unsigned int* pp = pLdsU + n * 16;     // row = lane%16
            for (int j = 0; j < 8; ++j) aP.u[j] = pp[a_pair_idx(j, sel)];
        }
        __syncthreads();

#ifdef HAVE_ASYNC_LDS
        // V B-tiles from LDS: element = vTile[nt*16+n][kk], contiguous.
        for (int nt = 0; nt < 4; ++nt) {
            const unsigned int* vp = (const unsigned int*)
                (&vTile[buf][(nt * 16 + n) * 32 + sel * 16]);
            for (int j = 0; j < 8; ++j) bV[nt].u[j] = vp[j];
        }
#else
        // V B-tiles from transposed V in global: contiguous per lane.
        for (int nt = 0; nt < 4; ++nt) {
            const unsigned int* vp = (const unsigned int*)
                (Vb + (size_t)(nt * 16 + n) * TLEN + s0 + sel * 16);
            for (int j = 0; j < 8; ++j) bV[nt].u[j] = vp[j];
        }
#endif
        for (int r = 0; r < 8; ++r) {
            acc0[r] *= fac[r]; acc1[r] *= fac[r];
            acc2[r] *= fac[r]; acc3[r] *= fac[r];
        }
        acc0 = __builtin_amdgcn_wmma_f32_16x16x32_f16(false, aP.h, false, bV[0].h, (short)0, acc0, false, false);
        acc1 = __builtin_amdgcn_wmma_f32_16x16x32_f16(false, aP.h, false, bV[1].h, (short)0, acc1, false, false);
        acc2 = __builtin_amdgcn_wmma_f32_16x16x32_f16(false, aP.h, false, bV[2].h, (short)0, acc2, false, false);
        acc3 = __builtin_amdgcn_wmma_f32_16x16x32_f16(false, aP.h, false, bV[3].h, (short)0, acc3, false, false);
    }

    for (int r = 0; r < 8; ++r) {
        const int row = rowbase + r + sel * 8;
        const float inv = 1.0f / lrow[r];
        float* op = out + ((size_t)b * TLEN + row) * HDIM;
        op[n]      = acc0[r] * inv;
        op[16 + n] = acc1[r] * inv;
        op[32 + n] = acc2[r] * inv;
        op[48 + n] = acc3[r] * inv;
    }
}

extern "C" void kernel_launch(void* const* d_in, const int* in_sizes, int n_in,
                              void* d_out, int out_size, void* d_ws, size_t ws_size,
                              hipStream_t stream) {
    (void)in_sizes; (void)n_in; (void)out_size; (void)ws_size;
    const float* x  = (const float*)d_in[0];
    const float* Wk = (const float*)d_in[1];
    const float* Wq = (const float*)d_in[2];
    const float* Wv = (const float*)d_in[3];
    float* out = (float*)d_out;

    // workspace layout (f16): Qr | Kr | Vt | Wc(packed uint)
    const size_t nqkv = (size_t)BATCH * TLEN * HDIM;     // 1M elements each
    _Float16* Qr = (_Float16*)d_ws;
    _Float16* Kr = Qr + nqkv;
    _Float16* Vt = Kr + nqkv;
    unsigned int* Wc = (unsigned int*)(Vt + nqkv);       // 3*512*64 f16

    head_wconv<<<dim3(16, 3), 128, 0, stream>>>(Wk, Wq, Wv, Wc);
    head_proj<<<dim3((BATCH * TLEN) / 16), 32, 0, stream>>>(x, Wc, Qr, Kr, Vt);
    head_attn<<<dim3(TLEN / 16, BATCH), 32, 0, stream>>>(Qr, Kr, Vt, out);
}
